// GraphCast_76390288327355
// MI455X (gfx1250) — matrix operand order repaired and checked by reading, not compiled, not run on hardware
//
#include <hip/hip_runtime.h>
#include <hip/hip_bf16.h>

// ---------------------------------------------------------------------------
// GraphCast g2m block on gfx1250: bf16 WMMA GEMMs with f32 accumulation.
// 32-row M tiles (2x16 per wave) to amortize B-fragment loads over 4 WMMAs.
// ---------------------------------------------------------------------------

typedef __attribute__((ext_vector_type(16))) __bf16 v16bf;
typedef __attribute__((ext_vector_type(8)))  __bf16 v8bf;
typedef __attribute__((ext_vector_type(8)))  float  v8f;

#define Hdim   256
#define E_TOT  260640
#define NG_TOT 65160
#define NM_TOT 40962

// padded LDS strides (in elements); all row offsets stay 16B aligned
#define HSTRIDE 264   // bf16, 256 + 8
#define OSTRIDE 260   // f32,  256 + 4

// ---------------------------------------------------------------------------
// WMMA helpers
// ---------------------------------------------------------------------------
__device__ __forceinline__ v8f wmma_bf16(v16bf a, v16bf b, v8f c) {
  // D = A(16x32 bf16) * B(32x16 bf16) + C(16x16 f32)
  return __builtin_amdgcn_wmma_f32_16x16x32_bf16(
      /*neg_a=*/false, a, /*neg_b=*/false, b,
      /*c_mod=*/(short)0, c, /*reuse_a=*/false, /*reuse_b=*/false);
}

// A-fragment (16x32 bf16) from row-major LDS tile.
// lane l: row = l%16, half = l/16.
//   elems 0..15 -> K = k0 + 16*half + 0..15
__device__ __forceinline__ v16bf load_a_frag(const __bf16* rowbase, int k0,
                                             int halfsel) {
  const v8bf lo = *(const v8bf*)(rowbase + k0 + (halfsel << 4));
  const v8bf hi = *(const v8bf*)(rowbase + k0 + 8 + (halfsel << 4));
  v16bf a;
#pragma unroll
  for (int i = 0; i < 8; ++i) { a[i] = lo[i]; a[i + 8] = hi[i]; }
  return a;
}

// ---------------------------------------------------------------------------
// Fused  Linear -> SiLU -> Linear -> LayerNorm  on one (16*MT)-row tile.
//  - blockDim.x == 256 (8 waves); wave w computes output columns 32w..32w+31
//    for all MT row sub-tiles (2*MT accumulators per lane).
//  - sA: bf16 input tile [16*MT x KDIM], row stride ASTRIDE.
//  - W1f/W2f: pre-swizzled bf16 B-fragment blobs (frag = kt*16+nt, 512 elems,
//    lane-major: lane*16+e, K = kt*32 + 16*(lane/16) + e, N = nt*16+lane%16).
//  - Result (post-LayerNorm, f32) left in sOut [16*MT x OSTRIDE].
// ---------------------------------------------------------------------------
template <int KDIM, int ASTRIDE, int MT>
__device__ __forceinline__ void mlp_tile(
    const __bf16* __restrict__ sA,
    const __bf16* __restrict__ W1f, const float* __restrict__ b1,
    const __bf16* __restrict__ W2f, const float* __restrict__ b2,
    const float* __restrict__ gam, const float* __restrict__ bet,
    __bf16* __restrict__ sH, float* __restrict__ sOut,
    float* __restrict__ sRed) {
  constexpr int MROWS = 16 * MT;
  const int tid = threadIdx.x;
  const int wave = tid >> 5;
  const int lane = tid & 31;
  const int halfsel = (lane >> 4) & 1;
  const int mrow = lane & 15;
  const int ncol = (wave << 5) + (lane & 15);  // tile0 column; tile1 = +16

  // -------- GEMM1: [MROWS,KDIM] x [KDIM,256] --------
  v8f acc[MT][2] = {};
  {
    constexpr int KT1 = KDIM >> 5;
#pragma unroll
    for (int kt = 0; kt < KT1; ++kt) {
      const __bf16* fb = W1f + ((size_t)(kt * 16 + (wave << 1)) << 9);
      if (kt + 1 < KT1)
        __builtin_prefetch(W1f + ((size_t)((kt + 1) * 16 + (wave << 1)) << 9),
                           0, 0);
      const v16bf b0 = *(const v16bf*)(fb + (lane << 4));
      const v16bf bv = *(const v16bf*)(fb + 512 + (lane << 4));
#pragma unroll
      for (int mt = 0; mt < MT; ++mt) {
        const v16bf a =
            load_a_frag(sA + (mrow + 16 * mt) * ASTRIDE, kt << 5, halfsel);
        acc[mt][0] = wmma_bf16(a, b0, acc[mt][0]);
        acc[mt][1] = wmma_bf16(a, bv, acc[mt][1]);
      }
    }
  }
  // bias + SiLU -> bf16 hidden tile
  {
    const float bia0 = b1[ncol];
    const float bia1 = b1[ncol + 16];
#pragma unroll
    for (int mt = 0; mt < MT; ++mt) {
#pragma unroll
      for (int r = 0; r < 8; ++r) {
        const int m = 16 * mt + r + (halfsel << 3);
        float x0 = acc[mt][0][r] + bia0;
        float x1 = acc[mt][1][r] + bia1;
        x0 = x0 / (1.0f + __expf(-x0));
        x1 = x1 / (1.0f + __expf(-x1));
        sH[m * HSTRIDE + ncol]      = (__bf16)x0;
        sH[m * HSTRIDE + ncol + 16] = (__bf16)x1;
      }
    }
  }
  __syncthreads();

  // -------- GEMM2: [MROWS,256] x [256,256] --------
  v8f dcc[MT][2] = {};
  {
#pragma unroll
    for (int kt = 0; kt < 8; ++kt) {
      const __bf16* fb = W2f + ((size_t)(kt * 16 + (wave << 1)) << 9);
      if (kt + 1 < 8)
        __builtin_prefetch(W2f + ((size_t)((kt + 1) * 16 + (wave << 1)) << 9),
                           0, 0);
      const v16bf b0 = *(const v16bf*)(fb + (lane << 4));
      const v16bf bv = *(const v16bf*)(fb + 512 + (lane << 4));
#pragma unroll
      for (int mt = 0; mt < MT; ++mt) {
        const v16bf a =
            load_a_frag(sH + (mrow + 16 * mt) * HSTRIDE, kt << 5, halfsel);
        dcc[mt][0] = wmma_bf16(a, b0, dcc[mt][0]);
        dcc[mt][1] = wmma_bf16(a, bv, dcc[mt][1]);
      }
    }
  }
  // bias -> f32 output tile
  {
    const float bia0 = b2[ncol];
    const float bia1 = b2[ncol + 16];
#pragma unroll
    for (int mt = 0; mt < MT; ++mt) {
#pragma unroll
      for (int r = 0; r < 8; ++r) {
        const int m = 16 * mt + r + (halfsel << 3);
        sOut[m * OSTRIDE + ncol]      = dcc[mt][0][r] + bia0;
        sOut[m * OSTRIDE + ncol + 16] = dcc[mt][1][r] + bia1;
      }
    }
  }
  __syncthreads();

  // -------- LayerNorm over each row of 256 --------
  const int sub = tid & 15;
#pragma unroll
  for (int rr = (tid >> 4); rr < MROWS; rr += 16) {
    const float* orow = sOut + rr * OSTRIDE + (sub << 4);
    float s = 0.0f, q = 0.0f;
#pragma unroll
    for (int i = 0; i < 16; ++i) {
      const float v = orow[i];
      s += v;
      q += v * v;
    }
    sRed[rr * 16 + sub]                = s;
    sRed[MROWS * 16 + rr * 16 + sub]   = q;
  }
  __syncthreads();
#pragma unroll
  for (int rr = (tid >> 4); rr < MROWS; rr += 16) {
    float su = 0.0f, qu = 0.0f;
#pragma unroll
    for (int i = 0; i < 16; ++i) {
      su += sRed[rr * 16 + i];
      qu += sRed[MROWS * 16 + rr * 16 + i];
    }
    const float mu   = su * (1.0f / 256.0f);
    const float var  = qu * (1.0f / 256.0f) - mu * mu;
    const float rstd = rsqrtf(var + 1e-5f);
    float* orow = sOut + rr * OSTRIDE + (sub << 4);
#pragma unroll
    for (int i = 0; i < 16; ++i) {
      const int c = (sub << 4) + i;
      orow[i] = (orow[i] - mu) * rstd * gam[c] + bet[c];
    }
  }
  __syncthreads();
}

// ---------------------------------------------------------------------------
// Prep kernels
// ---------------------------------------------------------------------------
__global__ void zero_f32(float* __restrict__ p, long n) {
  long i = (long)blockIdx.x * blockDim.x + threadIdx.x;
  const long stride = (long)gridDim.x * blockDim.x;
  for (; i < n; i += stride) p[i] = 0.0f;
}

// W [K,256] f32 row-major -> bf16 B-fragment blobs (one wave per 32x16 frag)
__global__ void convert_w(const float* __restrict__ W, __bf16* __restrict__ out,
                          int ktiles) {
  const int frag = blockIdx.x * (blockDim.x >> 5) + (threadIdx.x >> 5);
  const int lane = threadIdx.x & 31;
  if (frag >= ktiles * 16) return;
  const int kt = frag >> 4;
  const int nt = frag & 15;
  const int halfsel = lane >> 4;
  const int n = (nt << 4) + (lane & 15);
  __bf16* dst = out + ((size_t)frag << 9) + (lane << 4);
#pragma unroll
  for (int e = 0; e < 16; ++e) {
    const int k = (kt << 5) + (halfsel << 4) + e;
    dst[e] = (__bf16)W[(size_t)k * Hdim + n];
  }
}

// ---------------------------------------------------------------------------
// Main kernels (32-row tiles)
// ---------------------------------------------------------------------------
// Edge MLP + scatter-sum into agg. 32 edges per block (E % 32 == 0).
__global__ __launch_bounds__(256) void edge_kernel(
    const float* __restrict__ efeat, const float* __restrict__ grid,
    const float* __restrict__ mesh, const int* __restrict__ src_idx,
    const int* __restrict__ dst_idx, const __bf16* __restrict__ W1f,
    const float* __restrict__ b1, const __bf16* __restrict__ W2f,
    const float* __restrict__ b2, const float* __restrict__ gam,
    const float* __restrict__ bet, float* __restrict__ agg) {
  __shared__ __bf16 sA[32 * 776];  // [32 x 768] bf16 cat tile (padded)
  __shared__ __bf16 sH[32 * HSTRIDE];
  __shared__ float  sOut[32 * OSTRIDE];
  __shared__ float  sRed[1024];

  const int tid = threadIdx.x;
  const int cb  = (tid & 15) << 4;

  for (int rr = (tid >> 4); rr < 32; rr += 16) {
    const size_t eidx = (size_t)blockIdx.x * 32 + rr;
    const int si = src_idx[eidx];
    const int di = dst_idx[eidx];
    const float* p0 = efeat + eidx * Hdim + cb;
    const float* p1 = grid + (size_t)si * Hdim + cb;
    const float* p2 = mesh + (size_t)di * Hdim + cb;
    __bf16* arow = sA + rr * 776;
#pragma unroll
    for (int i = 0; i < 16; ++i) arow[cb + i] = (__bf16)p0[i];
#pragma unroll
    for (int i = 0; i < 16; ++i) arow[256 + cb + i] = (__bf16)p1[i];
#pragma unroll
    for (int i = 0; i < 16; ++i) arow[512 + cb + i] = (__bf16)p2[i];
  }
  __syncthreads();

  mlp_tile<768, 776, 2>(sA, W1f, b1, W2f, b2, gam, bet, sH, sOut, sRed);

  // segment_sum: scatter-add LN output rows into agg[dst]
  for (int rr = (tid >> 4); rr < 32; rr += 16) {
    const size_t eidx = (size_t)blockIdx.x * 32 + rr;
    const int di = dst_idx[eidx];
    float* dst = agg + (size_t)di * Hdim + cb;
    const float* orow = sOut + rr * OSTRIDE + cb;
#pragma unroll
    for (int i = 0; i < 16; ++i) atomicAdd(&dst[i], orow[i]);
  }
}

// Mesh-node MLP on [agg | mesh] + residual. 32 nodes per block (tail-guarded).
__global__ __launch_bounds__(256) void mesh_kernel(
    const float* __restrict__ agg, const float* __restrict__ mesh,
    const __bf16* __restrict__ W1f, const float* __restrict__ b1,
    const __bf16* __restrict__ W2f, const float* __restrict__ b2,
    const float* __restrict__ gam, const float* __restrict__ bet,
    float* __restrict__ out) {
  __shared__ __bf16 sA[32 * 520];  // [32 x 512] bf16 (padded)
  __shared__ __bf16 sH[32 * HSTRIDE];
  __shared__ float  sOut[32 * OSTRIDE];
  __shared__ float  sRed[1024];

  const int tid = threadIdx.x;
  const int cb  = (tid & 15) << 4;

  for (int rr = (tid >> 4); rr < 32; rr += 16) {
    const size_t node = (size_t)blockIdx.x * 32 + rr;
    __bf16* arow = sA + rr * 520;
    if (node < (size_t)NM_TOT) {
      const float* p0 = agg + node * Hdim + cb;
      const float* p1 = mesh + node * Hdim + cb;
#pragma unroll
      for (int i = 0; i < 16; ++i) arow[cb + i] = (__bf16)p0[i];
#pragma unroll
      for (int i = 0; i < 16; ++i) arow[256 + cb + i] = (__bf16)p1[i];
    } else {
#pragma unroll
      for (int i = 0; i < 16; ++i) arow[cb + i] = (__bf16)0.0f;
#pragma unroll
      for (int i = 0; i < 16; ++i) arow[256 + cb + i] = (__bf16)0.0f;
    }
  }
  __syncthreads();

  mlp_tile<512, 520, 2>(sA, W1f, b1, W2f, b2, gam, bet, sH, sOut, sRed);

  for (int rr = (tid >> 4); rr < 32; rr += 16) {
    const size_t node = (size_t)blockIdx.x * 32 + rr;
    if (node < (size_t)NM_TOT) {
      const float* orow = sOut + rr * OSTRIDE + cb;
      const float* mrow = mesh + node * Hdim + cb;
      float* drow = out + node * Hdim + cb;
#pragma unroll
      for (int i = 0; i < 16; ++i) drow[i] = mrow[i] + orow[i];
    }
  }
}

// Grid-node MLP + residual. 32 nodes per block (tail-guarded).
__global__ __launch_bounds__(256) void grid_kernel(
    const float* __restrict__ grid, const __bf16* __restrict__ W1f,
    const float* __restrict__ b1, const __bf16* __restrict__ W2f,
    const float* __restrict__ b2, const float* __restrict__ gam,
    const float* __restrict__ bet, float* __restrict__ out) {
  __shared__ __bf16 sA[32 * HSTRIDE];  // [32 x 256] bf16 (padded)
  __shared__ __bf16 sH[32 * HSTRIDE];
  __shared__ float  sOut[32 * OSTRIDE];
  __shared__ float  sRed[1024];

  const int tid = threadIdx.x;
  const int cb  = (tid & 15) << 4;

  for (int rr = (tid >> 4); rr < 32; rr += 16) {
    const size_t node = (size_t)blockIdx.x * 32 + rr;
    __bf16* arow = sA + rr * HSTRIDE;
    if (node < (size_t)NG_TOT) {
      const float* p0 = grid + node * Hdim + cb;
#pragma unroll
      for (int i = 0; i < 16; ++i) arow[cb + i] = (__bf16)p0[i];
    } else {
#pragma unroll
      for (int i = 0; i < 16; ++i) arow[cb + i] = (__bf16)0.0f;
    }
  }
  __syncthreads();

  mlp_tile<256, HSTRIDE, 2>(sA, W1f, b1, W2f, b2, gam, bet, sH, sOut, sRed);

  for (int rr = (tid >> 4); rr < 32; rr += 16) {
    const size_t node = (size_t)blockIdx.x * 32 + rr;
    if (node < (size_t)NG_TOT) {
      const float* orow = sOut + rr * OSTRIDE + cb;
      const float* grow = grid + node * Hdim + cb;
      float* drow = out + node * Hdim + cb;
#pragma unroll
      for (int i = 0; i < 16; ++i) drow[i] = grow[i] + orow[i];
    }
  }
}

// ---------------------------------------------------------------------------
// Launch
// ---------------------------------------------------------------------------
extern "C" void kernel_launch(void* const* d_in, const int* in_sizes, int n_in,
                              void* d_out, int out_size, void* d_ws,
                              size_t ws_size, hipStream_t stream) {
  (void)in_sizes; (void)n_in; (void)out_size; (void)ws_size;

  const float* efeat  = (const float*)d_in[0];
  const float* grid   = (const float*)d_in[1];
  const float* mesh   = (const float*)d_in[2];
  const int* src_idx  = (const int*)d_in[3];
  const int* dst_idx  = (const int*)d_in[4];
  const float* eW1 = (const float*)d_in[5];
  const float* eb1 = (const float*)d_in[6];
  const float* eW2 = (const float*)d_in[7];
  const float* eb2 = (const float*)d_in[8];
  const float* eg  = (const float*)d_in[9];
  const float* ebt = (const float*)d_in[10];
  const float* sW1 = (const float*)d_in[11];
  const float* sb1 = (const float*)d_in[12];
  const float* sW2 = (const float*)d_in[13];
  const float* sb2 = (const float*)d_in[14];
  const float* sg  = (const float*)d_in[15];
  const float* sbt = (const float*)d_in[16];
  const float* dW1 = (const float*)d_in[17];
  const float* db1 = (const float*)d_in[18];
  const float* dW2 = (const float*)d_in[19];
  const float* db2 = (const float*)d_in[20];
  const float* dg  = (const float*)d_in[21];
  const float* dbt = (const float*)d_in[22];

  float* out_grid = (float*)d_out;
  float* out_mesh = out_grid + (size_t)NG_TOT * Hdim;

  // workspace layout: [ agg f32 NM*H | bf16 weight fragment blobs ]
  float* agg = (float*)d_ws;
  __bf16* fr = (__bf16*)((char*)d_ws + (size_t)NM_TOT * Hdim * sizeof(float));
  __bf16* eW1f = fr;                       // 24 ktiles * 16 ntiles * 512
  __bf16* eW2f = eW1f + (size_t)24 * 16 * 512;
  __bf16* dW1f = eW2f + (size_t)8 * 16 * 512;
  __bf16* dW2f = dW1f + (size_t)16 * 16 * 512;
  __bf16* sW1f = dW2f + (size_t)8 * 16 * 512;
  __bf16* sW2f = sW1f + (size_t)8 * 16 * 512;

  zero_f32<<<2048, 256, 0, stream>>>(agg, (long)NM_TOT * Hdim);
  convert_w<<<(24 * 16 + 7) / 8, 256, 0, stream>>>(eW1, eW1f, 24);
  convert_w<<<(8 * 16 + 7) / 8, 256, 0, stream>>>(eW2, eW2f, 8);
  convert_w<<<(16 * 16 + 7) / 8, 256, 0, stream>>>(dW1, dW1f, 16);
  convert_w<<<(8 * 16 + 7) / 8, 256, 0, stream>>>(dW2, dW2f, 8);
  convert_w<<<(8 * 16 + 7) / 8, 256, 0, stream>>>(sW1, sW1f, 8);
  convert_w<<<(8 * 16 + 7) / 8, 256, 0, stream>>>(sW2, sW2f, 8);

  edge_kernel<<<E_TOT / 32, 256, 0, stream>>>(efeat, grid, mesh, src_idx,
                                              dst_idx, eW1f, eb1, eW2f, eb2,
                                              eg, ebt, agg);
  mesh_kernel<<<(NM_TOT + 31) / 32, 256, 0, stream>>>(agg, mesh, dW1f, db1,
                                                      dW2f, db2, dg, dbt,
                                                      out_mesh);
  grid_kernel<<<(NG_TOT + 31) / 32, 256, 0, stream>>>(grid, sW1f, sb1, sW2f,
                                                      sb2, sg, sbt, out_grid);
}
